// MambaDynamicsModel_17403207483676
// MI455X (gfx1250) — matrix operand order repaired and checked by reading, not compiled, not run on hardware
//
#include <hip/hip_runtime.h>
#include <hip/hip_bf16.h>

// ---------------------------------------------------------------------------
// Mamba dynamics model on gfx1250:
//   - weights pre-converted to bf16 and transposed to N-major (one-shot)
//   - 4 bf16-WMMA GEMMs with double-buffered LDS fed by
//     global_load_async_to_lds_b128 (ASYNCcnt pipeline)
//   - fused pointwise + RMS-norm kernel
// ---------------------------------------------------------------------------

typedef __attribute__((ext_vector_type(16))) __bf16 v16bf;
typedef __attribute__((ext_vector_type(8)))  float  v8f;

#define BM 128
#define BN 128
#define BK 32
#define LDK 40   // padded LDS row stride in bf16 elems (80 B, 16-B aligned)

union FragBf16 {
    v16bf v;
    uint4 q[2];
};

__device__ __forceinline__ float sigmoid_f(float x) { return 1.0f / (1.0f + __expf(-x)); }
__device__ __forceinline__ float silu_f(float x)    { return x * sigmoid_f(x); }

// --- CDNA5 async global->LDS copy (16 B per lane), tracked by ASYNCcnt ----
__device__ __forceinline__ void async_b128(unsigned lds_off, const void* gaddr) {
    asm volatile("global_load_async_to_lds_b128 %0, %1, off"
                 :: "v"(lds_off), "v"(gaddr) : "memory");
}
__device__ __forceinline__ void wait_async0() {
    asm volatile("s_wait_asynccnt 0x0" ::: "memory");
}

// ---------------------------------------------------------------------------
// Tiled GEMM: C[M,N] = A[M,K](bf16, row-major) * Bt[N,K](bf16, N-major) (+bias)
// 256 threads = 8 waves; each wave computes a 32x64 tile (2x4 WMMA tiles).
// Double-buffered LDS, async copies overlap WMMA of the previous K-step.
// ---------------------------------------------------------------------------
template <bool OUT_BF16, bool HAS_BIAS>
__global__ __launch_bounds__(256) void gemm_bf16_wmma(
    const __bf16* __restrict__ A,     // M x K bf16
    const __bf16* __restrict__ Bt,    // N x K bf16 (pre-transposed weight)
    const float*  __restrict__ bias,  // N (may be null)
    void* __restrict__ Cout,          // M x N, bf16 or f32
    int M, int N, int K)
{
    __shared__ __align__(16) __bf16 As[2][BM][LDK];
    __shared__ __align__(16) __bf16 Bs[2][BN][LDK];

    const int tid  = threadIdx.x;
    const int wave = tid >> 5;
    const int lane = tid & 31;
    const int lr   = lane & 15;
    const int hi   = lane >> 4;

    const int m0 = blockIdx.y * BM;
    const int n0 = blockIdx.x * BN;

    const int wm = (wave >> 1) * 32;  // wave row: 0,32,64,96
    const int wn = (wave & 1) * 64;   // wave col: 0,64

    const unsigned aBase = (unsigned)(uintptr_t)&As[0][0][0];
    const unsigned bBase = (unsigned)(uintptr_t)&Bs[0][0][0];
    const unsigned bufStrideA = (unsigned)(BM * LDK * 2);
    const unsigned bufStrideB = (unsigned)(BN * LDK * 2);

    // per-thread copy assignment: 512 16-B chunks per tile, 2 per thread
    const int cRow0 = tid >> 1;                 // unused split kept simple below

    v8f acc[2][4];
#pragma unroll
    for (int i = 0; i < 2; ++i)
#pragma unroll
        for (int j = 0; j < 4; ++j)
#pragma unroll
            for (int e = 0; e < 8; ++e) acc[i][j][e] = 0.0f;

    auto issue_tile = [&](int kb, int buf) {
        const unsigned aB = aBase + (unsigned)buf * bufStrideA;
        const unsigned bB = bBase + (unsigned)buf * bufStrideB;
#pragma unroll
        for (int c0 = 0; c0 < 2; ++c0) {
            const int c   = tid + c0 * 256;     // 0..511
            const int row = c >> 2;             // 0..127
            const int ko  = (c & 3) * 8;        // 0,8,16,24 (bf16 elems)
            async_b128(aB + (unsigned)(row * LDK + ko) * 2u,
                       A + (size_t)(m0 + row) * K + kb + ko);
        }
#pragma unroll
        for (int c0 = 0; c0 < 2; ++c0) {
            const int c   = tid + c0 * 256;
            const int row = c >> 2;
            const int ko  = (c & 3) * 8;
            if (n0 + row < N)
                async_b128(bB + (unsigned)(row * LDK + ko) * 2u,
                           Bt + (size_t)(n0 + row) * K + kb + ko);
        }
    };

    const int T = K / BK;
    issue_tile(0, 0);

    for (int it = 0; it < T; ++it) {
        wait_async0();          // own wave's copies for tile `it` landed
        __syncthreads();        // everyone's copies landed; prev compute done
        if (it + 1 < T) issue_tile((it + 1) * BK, (it + 1) & 1);

        const int buf = it & 1;

        FragBf16 afr[2], bfr[4];
#pragma unroll
        for (int mt = 0; mt < 2; ++mt) {
            const int m  = wm + mt * 16 + lr;
            const int kh = hi * 8;              // lanes 16-31 own K+8 halves
            afr[mt].q[0] = *(const uint4*)&As[buf][m][kh];
            afr[mt].q[1] = *(const uint4*)&As[buf][m][16 + kh];
        }
#pragma unroll
        for (int nt = 0; nt < 4; ++nt) {
            const int n  = wn + nt * 16 + lr;
            const int ko = hi * 8;              // lanes 16-31 own K+8 halves
            bfr[nt].q[0] = *(const uint4*)&Bs[buf][n][ko];
            bfr[nt].q[1] = *(const uint4*)&Bs[buf][n][16 + ko];
        }

#pragma unroll
        for (int mt = 0; mt < 2; ++mt)
#pragma unroll
            for (int nt = 0; nt < 4; ++nt)
                acc[mt][nt] = __builtin_amdgcn_wmma_f32_16x16x32_bf16(
                    false, afr[mt].v, false, bfr[nt].v,
                    (short)0, acc[mt][nt], false, false);
    }

    // ---- epilogue: VGPR r -> row hi*8+r, lane -> column ----
#pragma unroll
    for (int mt = 0; mt < 2; ++mt) {
#pragma unroll
        for (int nt = 0; nt < 4; ++nt) {
            const int gn = n0 + wn + nt * 16 + lr;
            if (gn >= N) continue;
            const float bv = HAS_BIAS ? bias[gn] : 0.0f;
#pragma unroll
            for (int r = 0; r < 8; ++r) {
                const int gm = m0 + wm + mt * 16 + hi * 8 + r;
                const float val = acc[mt][nt][r] + bv;
                if (OUT_BF16)
                    ((__bf16*)Cout)[(size_t)gm * N + gn] = (__bf16)val;
                else
                    ((float*)Cout)[(size_t)gm * N + gn] = val;
            }
        }
    }
    (void)cRow0;
}

// ---------------------------------------------------------------------------
// One-shot: W[K,N] f32  ->  Wt[N,K] bf16 (transpose + convert), 32x32 tiles.
// ---------------------------------------------------------------------------
__global__ __launch_bounds__(256) void transpose_cvt_kernel(
    const float* __restrict__ W, __bf16* __restrict__ Wt, int K, int N)
{
    __shared__ float t[32][33];
    const int nb = blockIdx.x * 32;
    const int kb = blockIdx.y * 32;
    const int tx = threadIdx.x & 31;
    const int ty = threadIdx.x >> 5;   // 0..7
#pragma unroll
    for (int i = 0; i < 4; ++i) {
        const int k = kb + ty + i * 8;
        const int n = nb + tx;
        t[ty + i * 8][tx] = (k < K && n < N) ? W[(size_t)k * N + n] : 0.0f;
    }
    __syncthreads();
#pragma unroll
    for (int i = 0; i < 4; ++i) {
        const int n = nb + ty + i * 8;
        const int k = kb + tx;
        if (n < N && k < K)
            Wt[(size_t)n * K + k] = (__bf16)t[tx][ty + i * 8];
    }
}

// ---------------------------------------------------------------------------
// Pack concat(state, action) -> bf16 X [16384 x 320]
// ---------------------------------------------------------------------------
__global__ __launch_bounds__(256) void pack_x_kernel(
    const float* __restrict__ state, const float* __restrict__ action,
    __bf16* __restrict__ X)
{
    const int i = blockIdx.x * 256 + threadIdx.x;
    const int total = 16384 * 320;
    if (i >= total) return;
    const int row = i / 320;
    const int col = i - row * 320;
    const float v = (col < 256) ? state[row * 256 + col]
                                : action[row * 64 + (col - 256)];
    X[i] = (__bf16)v;
}

// ---------------------------------------------------------------------------
// Fused pointwise: conv-scale + SiLU, bc = sum(B*C), softplus(dt), head scale,
// gate with SiLU(z), RMS-norm. One block per row. Emits y bf16 [16384 x 2048].
// ---------------------------------------------------------------------------
__global__ __launch_bounds__(256) void pointwise_kernel(
    const float* __restrict__ zxbcdt,   // 16384 x 4256
    const float* __restrict__ conv_w,   // 2176 x 4 (use col 3)
    const float* __restrict__ conv_b,   // 2176
    const float* __restrict__ dt_bias,  // 32
    const float* __restrict__ Dp,       // 32
    const float* __restrict__ norm_w,   // 2048
    __bf16* __restrict__ Y)             // 16384 x 2048
{
    const int r   = blockIdx.x;
    const int tid = threadIdx.x;
    const float* zr  = zxbcdt + (size_t)r * 4256;
    const float* xBC = zr + 2048;

    __shared__ float sy[2048];
    __shared__ float red[256];
    __shared__ float sdt[32];
    __shared__ float sbc;

    float p = 0.0f;
    if (tid < 64) {
        const int cb = 2048 + tid;
        const int cc = 2112 + tid;
        const float bv = silu_f(xBC[cb] * conv_w[cb * 4 + 3] + conv_b[cb]);
        const float cv = silu_f(xBC[cc] * conv_w[cc * 4 + 3] + conv_b[cc]);
        p = bv * cv;
    }
    red[tid] = p;
    __syncthreads();
    for (int s = 128; s > 0; s >>= 1) {
        if (tid < s) red[tid] += red[tid + s];
        __syncthreads();
    }
    if (tid == 0) sbc = red[0];
    if (tid < 32) {
        const float d = zr[4224 + tid] + dt_bias[tid];
        sdt[tid] = (d > 20.0f) ? d : log1pf(__expf(d));
    }
    __syncthreads();

    const float bc = sbc;
    float ss = 0.0f;
    for (int c = tid; c < 2048; c += 256) {
        const int h = c >> 6;   // HEADDIM = 64
        const float xa = silu_f(xBC[c] * conv_w[c * 4 + 3] + conv_b[c]);
        const float scale = sdt[h] * bc + Dp[h];
        const float zv = zr[c];
        const float y = xa * scale * (zv * sigmoid_f(zv));
        sy[c] = y;
        ss += y * y;
    }
    red[tid] = ss;
    __syncthreads();
    for (int s = 128; s > 0; s >>= 1) {
        if (tid < s) red[tid] += red[tid + s];
        __syncthreads();
    }
    const float rstd = rsqrtf(red[0] * (1.0f / 2048.0f) + 1e-5f);

    __bf16* yr = Y + (size_t)r * 2048;
    for (int c = tid; c < 2048; c += 256) {
        yr[c] = (__bf16)(sy[c] * rstd * norm_w[c]);
    }
}

// ---------------------------------------------------------------------------
// Host launcher
// ---------------------------------------------------------------------------
extern "C" void kernel_launch(void* const* d_in, const int* in_sizes, int n_in,
                              void* d_out, int out_size, void* d_ws, size_t ws_size,
                              hipStream_t stream) {
    const float* state    = (const float*)d_in[0];
    const float* action   = (const float*)d_in[1];
    const float* W_in     = (const float*)d_in[2];
    const float* b_in     = (const float*)d_in[3];
    const float* W_inproj = (const float*)d_in[4];
    const float* conv_w   = (const float*)d_in[5];
    const float* conv_b   = (const float*)d_in[6];
    const float* dt_bias  = (const float*)d_in[7];
    // d_in[8] = A_log (unused by reference math)
    const float* Dp       = (const float*)d_in[9];
    const float* norm_w   = (const float*)d_in[10];
    const float* W_outproj= (const float*)d_in[11];
    const float* W_out    = (const float*)d_in[12];
    const float* b_out    = (const float*)d_in[13];

    char* ws = (char*)d_ws;
    // region A (off 0): X bf16 [16384x320], later reused as zxbcdt f32 [16384x4256]
    // region B: x0 bf16 [16384x1024], later reused as t bf16
    // region C: y bf16 [16384x2048]
    // region W: transposed bf16 weights
    const size_t offA  = 0;
    const size_t offB  = (size_t)16384 * 4256 * 4;          // 278,921,216
    const size_t offC  = offB + (size_t)16384 * 1024 * 2;   // +33,554,432
    const size_t offW0 = offC + (size_t)16384 * 2048 * 2;   // +67,108,864
    const size_t offW1 = offW0 + (size_t)1024 * 320 * 2;    // Wt_in
    const size_t offW2 = offW1 + (size_t)4256 * 1024 * 2;   // Wt_inproj
    const size_t offW3 = offW2 + (size_t)1024 * 2048 * 2;   // Wt_outproj

    __bf16* Xb      = (__bf16*)(ws + offA);
    float*  zxbcdt  = (float*)(ws + offA);
    __bf16* x0b     = (__bf16*)(ws + offB);
    __bf16* tb      = (__bf16*)(ws + offB);
    __bf16* Yb      = (__bf16*)(ws + offC);
    __bf16* WtIn    = (__bf16*)(ws + offW0);   // 1024 x 320
    __bf16* WtInPr  = (__bf16*)(ws + offW1);   // 4256 x 1024
    __bf16* WtOutPr = (__bf16*)(ws + offW2);   // 1024 x 2048
    __bf16* WtOut   = (__bf16*)(ws + offW3);   // 256 x 1024
    float*  out     = (float*)d_out;

    const int M = 16384;

    // 0) one-shot weight transpose+convert (f32 KxN -> bf16 NxK)
    transpose_cvt_kernel<<<dim3(1024 / 32, 320 / 32),  256, 0, stream>>>(W_in,      WtIn,    320, 1024);
    transpose_cvt_kernel<<<dim3(4256 / 32, 1024 / 32), 256, 0, stream>>>(W_inproj,  WtInPr, 1024, 4256);
    transpose_cvt_kernel<<<dim3(1024 / 32, 2048 / 32), 256, 0, stream>>>(W_outproj, WtOutPr,2048, 1024);
    transpose_cvt_kernel<<<dim3(256 / 32, 1024 / 32),  256, 0, stream>>>(W_out,     WtOut,  1024, 256);

    // 1) pack concat(state, action) -> bf16
    pack_x_kernel<<<(M * 320 + 255) / 256, 256, 0, stream>>>(state, action, Xb);

    // 2) x0 = X @ W_in + b_in   (M x 1024), store bf16
    gemm_bf16_wmma<true, true><<<dim3(1024 / BN, M / BM), 256, 0, stream>>>(
        Xb, WtIn, b_in, x0b, M, 1024, 320);

    // 3) zxbcdt = x0 @ W_inproj (M x 4256), store f32 (overlays dead X)
    gemm_bf16_wmma<false, false><<<dim3((4256 + BN - 1) / BN, M / BM), 256, 0, stream>>>(
        x0b, WtInPr, nullptr, zxbcdt, M, 4256, 1024);

    // 4) pointwise + RMS norm -> y bf16 (M x 2048)
    pointwise_kernel<<<M, 256, 0, stream>>>(zxbcdt, conv_w, conv_b, dt_bias,
                                            Dp, norm_w, Yb);

    // 5) t = y @ W_outproj (M x 1024), store bf16 (overlays dead x0)
    gemm_bf16_wmma<true, false><<<dim3(1024 / BN, M / BM), 256, 0, stream>>>(
        Yb, WtOutPr, nullptr, tb, M, 1024, 2048);

    // 6) out = t @ W_out + b_out (M x 256), f32 -> d_out
    gemm_bf16_wmma<false, true><<<dim3(256 / BN, M / BM), 256, 0, stream>>>(
        tb, WtOut, b_out, out, M, 256, 1024);
}
